// MultiModalAttention_66924180407350
// MI455X (gfx1250) — matrix operand order repaired and checked by reading, not compiled, not run on hardware
//
#include <hip/hip_runtime.h>

typedef __attribute__((ext_vector_type(16))) _Float16 v16h;
typedef __attribute__((ext_vector_type(8)))  float    v8f;
typedef unsigned int u32x4 __attribute__((ext_vector_type(4)));
typedef int          i32x4 __attribute__((ext_vector_type(4)));
typedef int          i32x8 __attribute__((ext_vector_type(8)));

constexpr int  Bc  = 2, Sc = 2048, Dc = 2048, Hc = 16, HDc = 128;
constexpr long NTOK  = (long)Bc * Sc;   // 4096 rows
constexpr long NELEM = NTOK * Dc;       // 8M elements
constexpr float SCALE = 0.08838834764831845f; // 1/sqrt(128)

// ---------- helpers ----------
__device__ inline v8f vzero8() {
  v8f z;
#pragma unroll
  for (int i = 0; i < 8; ++i) z[i] = 0.f;
  return z;
}

// A-fragment (16x32 f16): lane holds row M=lane%16; elems 0..7 <- K=kb..kb+7,
// elems 8..15 <- K=kb+16..kb+23, kb = 8*(lane>=16). Caller pre-offsets by kb.
__device__ inline v16h ldfragA(const _Float16* p) {
  union { v16h v; float4 f[2]; } u;
  u.f[0] = *(const float4*)(p);
  u.f[1] = *(const float4*)(p + 16);
  return u.v;
}
// B-fragment (32x16 f16): lane holds col N=lane%16; elems 0..7 <- K=kb..kb+7,
// elems 8..15 <- K=kb+16..kb+23, kb = 8*(lane>=16). Caller pre-offsets by kb.
__device__ inline v16h ldfragB(const _Float16* p) {
  union { v16h v; float4 f[2]; } u;
  u.f[0] = *(const float4*)(p);
  u.f[1] = *(const float4*)(p + 16);
  return u.v;
}

__device__ inline v8f wmma_f16(v16h a, v16h b, v8f c) {
  return __builtin_amdgcn_wmma_f32_16x16x32_f16(false, a, false, b,
                                                (short)0, c, false, false);
}

// LDS byte offset of a __shared__ object (flat LDS aperture keeps the
// allocation offset in addr[31:0]).
__device__ inline unsigned ldsoff(const void* p) {
  return (unsigned)(unsigned long long)(uintptr_t)p;
}

// TDM: async 2D tile load (tile_rows x tile_cols f16 elements) from a
// row-major tensor with row stride `stride` elements into contiguous LDS.
// D# packing per CDNA5 ISA.
__device__ inline void tdm_load_2d(unsigned lds_byte_off, const _Float16* gptr,
                                   unsigned tile_cols, unsigned tile_rows,
                                   unsigned td0, unsigned td1, unsigned stride) {
  unsigned long long ga = (unsigned long long)(uintptr_t)gptr;
  u32x4 g0;
  g0[0] = 1u;                                   // count=1, user descriptor
  g0[1] = lds_byte_off;                         // lds_addr
  g0[2] = (unsigned)(ga & 0xffffffffu);         // global_addr[31:0]
  g0[3] = (unsigned)((ga >> 32) & 0x01ffffffu)  // global_addr[56:32]
          | 0x80000000u;                        // type=2 ("image")
  i32x8 g1;
  g1[0] = 0x00010000;                           // data_size=1 -> 2 bytes
  g1[1] = (int)((td0 & 0xffffu) << 16);                       // tensor_dim0 lo
  g1[2] = (int)(((td0 >> 16) & 0xffffu) | ((td1 & 0xffffu) << 16)); // td0 hi | td1 lo
  g1[3] = (int)(((td1 >> 16) & 0xffffu) | ((tile_cols & 0xffffu) << 16)); // td1 hi | tile_dim0
  g1[4] = (int)(tile_rows & 0xffffu);           // tile_dim1 (tile_dim2=0)
  g1[5] = (int)stride;                          // tensor_dim0_stride[31:0]
  g1[6] = 0;                                    // stride hi | dim1_stride lo
  g1[7] = 0;
  i32x4 z4;
  z4[0] = 0; z4[1] = 0; z4[2] = 0; z4[3] = 0;
  i32x8 z8;
#pragma unroll
  for (int i = 0; i < 8; ++i) z8[i] = 0;
  __builtin_amdgcn_tensor_load_to_lds(g0, g1, z4, z4, z8, 0);
}

// ---------- elementwise / layout kernels ----------
__global__ void cvt_f32_f16(const float* __restrict__ in, _Float16* __restrict__ out, long n) {
  long i = (long)blockIdx.x * blockDim.x + threadIdx.x;
  if (i < n) out[i] = (_Float16)in[i];
}

// wt[n][k] = w[k][n], f32 -> f16
__global__ void transpose_to_f16(const float* __restrict__ w, _Float16* __restrict__ wt, int dim) {
  long i = (long)blockIdx.x * blockDim.x + threadIdx.x;
  if (i < (long)dim * dim) {
    int k = (int)(i / dim), n = (int)(i % dim);
    wt[(long)n * dim + k] = (_Float16)w[i];
  }
}

// Vt[b][h][hd][s] = V[b][s][h*HD+hd]
__global__ void transpose_v_kernel(const _Float16* __restrict__ V, _Float16* __restrict__ Vt) {
  long i = (long)blockIdx.x * blockDim.x + threadIdx.x;
  if (i < NELEM) {
    int  c   = (int)(i % Dc);
    long row = i / Dc;
    int  b = (int)(row / Sc), s = (int)(row % Sc);
    int  h = c / HDc, hd = c % HDc;
    Vt[(((long)(b * Hc + h)) * HDc + hd) * Sc + s] = V[i];
  }
}

// In-place RoPE on one (b,s,h) row per thread (reference semantics:
// x1=x[::2], x2=x[1::2]; out = [x1*cos - x2*sin, x1*sin + x2*cos]).
__global__ void rope_inplace(_Float16* __restrict__ Q, _Float16* __restrict__ Kq) {
  int id = blockIdx.x * blockDim.x + threadIdx.x;
  if (id >= Bc * Sc * Hc) return;
  int h = id % Hc;
  int s = (id / Hc) % Sc;
  int b = id / (Hc * Sc);
  long base = ((long)(b * Sc + s)) * Dc + h * HDc;

  float cs[64], sn[64];
#pragma unroll
  for (int j = 0; j < 64; ++j) {
    // inv_freq = 10000^(-2j/128) = exp(-j * ln(10000)/64)
    float ang = (float)s * __expf(-0.14391156f * (float)j);
    cs[j] = __cosf(ang);
    sn[j] = __sinf(ang);
  }
  float buf[128];
#pragma unroll
  for (int i = 0; i < 128; ++i) buf[i] = (float)Q[base + i];
#pragma unroll
  for (int j = 0; j < 64; ++j) {
    float x1 = buf[2 * j], x2 = buf[2 * j + 1];
    Q[base + j]      = (_Float16)(x1 * cs[j] - x2 * sn[j]);
    Q[base + 64 + j] = (_Float16)(x1 * sn[j] + x2 * cs[j]);
  }
#pragma unroll
  for (int i = 0; i < 128; ++i) buf[i] = (float)Kq[base + i];
#pragma unroll
  for (int j = 0; j < 64; ++j) {
    float x1 = buf[2 * j], x2 = buf[2 * j + 1];
    Kq[base + j]      = (_Float16)(x1 * cs[j] - x2 * sn[j]);
    Kq[base + 64 + j] = (_Float16)(x1 * sn[j] + x2 * cs[j]);
  }
}

// ---------- TDM-staged WMMA GEMM ----------
// C[M][N] = A[M][K](f16) * Bt[N][K](f16)^T + bias
// Block: 256 threads = 8 waves; block tile 64(M) x 256(N); wave tile 32x64.
// Per k-step (K+=32) wave 0 issues two TDM tile loads (A: 64x32, B: 256x32)
// into a double-buffered LDS stage; everyone syncs on TENSORcnt + barrier.
template <typename OutT>
__global__ void __launch_bounds__(256)
gemm_f16_wmma(const _Float16* __restrict__ A, const _Float16* __restrict__ Bt,
              const float* __restrict__ bias, OutT* __restrict__ C,
              int M, int N, int K) {
  __shared__ __attribute__((aligned(16))) _Float16 lsA[2][64][32];
  __shared__ __attribute__((aligned(16))) _Float16 lsB[2][256][32];

  const int lane   = threadIdx.x & 31;
  const int wave   = threadIdx.x >> 5;
  const int wm     = wave & 1;    // 0..1 : 32-row strip
  const int wn     = wave >> 1;   // 0..3 : 64-col strip
  const int l15    = lane & 15;
  const int half16 = lane >> 4;
  const int kb8    = half16 * 8;
  const int m0 = blockIdx.y * 64;
  const int n0 = blockIdx.x * 256;

  const _Float16* aG = A  + (long)m0 * K;
  const _Float16* bG = Bt + (long)n0 * K;

  v8f acc[2][4];
#pragma unroll
  for (int r = 0; r < 2; ++r)
#pragma unroll
    for (int j = 0; j < 4; ++j) acc[r][j] = vzero8();

  const int nsteps = K / 32;
  if (wave == 0) {
    tdm_load_2d(ldsoff(&lsA[0][0][0]), aG, 32, 64,  (unsigned)K, (unsigned)M, (unsigned)K);
    tdm_load_2d(ldsoff(&lsB[0][0][0]), bG, 32, 256, (unsigned)K, (unsigned)N, (unsigned)K);
  }

  for (int i = 0; i < nsteps; ++i) {
    const int bi = i & 1;
    if (wave == 0 && (i + 1) < nsteps) {
      const int nb = (i + 1) & 1;
      tdm_load_2d(ldsoff(&lsA[nb][0][0]), aG + (i + 1) * 32, 32, 64,
                  (unsigned)K, (unsigned)M, (unsigned)K);
      tdm_load_2d(ldsoff(&lsB[nb][0][0]), bG + (i + 1) * 32, 32, 256,
                  (unsigned)K, (unsigned)N, (unsigned)K);
    }
    if ((i + 1) < nsteps) __builtin_amdgcn_s_wait_tensorcnt(2);
    else                  __builtin_amdgcn_s_wait_tensorcnt(0);
    __syncthreads();

    v16h a0 = ldfragA(&lsA[bi][wm * 32 + l15][kb8]);
    v16h a1 = ldfragA(&lsA[bi][wm * 32 + 16 + l15][kb8]);
#pragma unroll
    for (int j = 0; j < 4; ++j) {
      v16h b = ldfragB(&lsB[bi][wn * 64 + 16 * j + l15][kb8]);
      acc[0][j] = wmma_f16(a0, b, acc[0][j]);
      acc[1][j] = wmma_f16(a1, b, acc[1][j]);
    }
    __syncthreads();
  }

#pragma unroll
  for (int r = 0; r < 2; ++r) {
#pragma unroll
    for (int j = 0; j < 4; ++j) {
      const int n  = n0 + wn * 64 + 16 * j + l15;
      const float bb = bias ? bias[n] : 0.f;
#pragma unroll
      for (int v = 0; v < 8; ++v) {
        const int m = m0 + wm * 32 + r * 16 + v + kb8;
        C[(long)m * N + n] = (OutT)(acc[r][j][v] + bb);
      }
    }
  }
}

// ---------- flash attention: one wave per 16-row q tile ----------
// The P transpose LDS buffer is wave-private and LDS ops are in-order per
// wave (DScnt), so no block barriers are needed in the k loop.
__global__ void __launch_bounds__(128)
attn_wmma(const _Float16* __restrict__ Q, const _Float16* __restrict__ Kh,
          const _Float16* __restrict__ Vt, const int* __restrict__ mask,
          _Float16* __restrict__ O) {
  __shared__ __attribute__((aligned(16))) _Float16 plds[4][16][40];
  const int lane   = threadIdx.x & 31;
  const int wave   = threadIdx.x >> 5;
  const int l15    = lane & 15;
  const int half16 = lane >> 4;
  const int kb8    = half16 * 8;

  const int tile = blockIdx.x * 4 + wave;
  const int qt = tile & (Sc / 16 - 1);
  const int h  = (tile / (Sc / 16)) & (Hc - 1);
  const int b  = tile / ((Sc / 16) * Hc);

  v16h aq[4];
  {
    const _Float16* qp = Q + ((long)(b * Sc + qt * 16 + l15)) * Dc + h * HDc + kb8;
#pragma unroll
    for (int c = 0; c < 4; ++c) aq[c] = ldfragA(qp + 32 * c);
  }

  v8f o[8];
#pragma unroll
  for (int j = 0; j < 8; ++j) o[j] = vzero8();
  float mrow[8], lrow[8];
#pragma unroll
  for (int v = 0; v < 8; ++v) { mrow[v] = -3.0e38f; lrow[v] = 0.f; }

  for (int k0 = 0; k0 < Sc; k0 += 32) {
    v8f s[2];
#pragma unroll
    for (int t = 0; t < 2; ++t) {
      s[t] = vzero8();
      const _Float16* kp = Kh + ((long)(b * Sc + k0 + 16 * t + l15)) * Dc + h * HDc + kb8;
#pragma unroll
      for (int c = 0; c < 4; ++c) {
        v16h bk = ldfragB(kp + 32 * c);
        s[t] = wmma_f16(aq[c], bk, s[t]);
      }
      const int mv = mask[b * Sc + k0 + 16 * t + l15];
#pragma unroll
      for (int v = 0; v < 8; ++v) {
        float xx = s[t][v] * SCALE;
        s[t][v] = (mv == 0) ? -1.0e9f : xx;
      }
    }
    float alpha[8];
#pragma unroll
    for (int v = 0; v < 8; ++v) {
      float mx = fmaxf(s[0][v], s[1][v]);
#pragma unroll
      for (int d = 8; d >= 1; d >>= 1) mx = fmaxf(mx, __shfl_xor(mx, d, 32));
      float mn = fmaxf(mrow[v], mx);
      alpha[v] = __expf(mrow[v] - mn);
      mrow[v] = mn;
      float p0 = __expf(s[0][v] - mn);
      float p1 = __expf(s[1][v] - mn);
      s[0][v] = p0; s[1][v] = p1;
      float rs = p0 + p1;
#pragma unroll
      for (int d = 8; d >= 1; d >>= 1) rs += __shfl_xor(rs, d, 32);
      lrow[v] = lrow[v] * alpha[v] + rs;
    }
#pragma unroll
    for (int j = 0; j < 8; ++j)
#pragma unroll
      for (int v = 0; v < 8; ++v) o[j][v] *= alpha[v];

    // P: C-layout -> A-fragment layout via wave-private LDS (in-order per wave)
#pragma unroll
    for (int t = 0; t < 2; ++t)
#pragma unroll
      for (int v = 0; v < 8; ++v)
        plds[wave][v + kb8][16 * t + l15] = (_Float16)s[t][v];
    v16h pa = ldfragA(&plds[wave][l15][kb8]);

    const _Float16* vp = Vt + ((long)((b * Hc + h) * HDc + l15)) * Sc + k0 + kb8;
#pragma unroll
    for (int j = 0; j < 8; ++j) {
      v16h bvf = ldfragB(vp + (long)16 * j * Sc);
      o[j] = wmma_f16(pa, bvf, o[j]);
    }
  }

#pragma unroll
  for (int j = 0; j < 8; ++j) {
#pragma unroll
    for (int v = 0; v < 8; ++v) {
      float r = o[j][v] / lrow[v];
      O[((long)(b * Sc + qt * 16 + v + kb8)) * Dc + h * HDc + 16 * j + l15] = (_Float16)r;
    }
  }
}

// ---------- launch ----------
extern "C" void kernel_launch(void* const* d_in, const int* in_sizes, int n_in,
                              void* d_out, int out_size, void* d_ws, size_t ws_size,
                              hipStream_t stream) {
  (void)in_sizes; (void)n_in; (void)out_size; (void)ws_size;
  const float* x  = (const float*)d_in[0];
  const int*   am = (const int*)  d_in[1];
  const float* wq = (const float*)d_in[2];
  const float* bq = (const float*)d_in[3];
  const float* wk = (const float*)d_in[4];
  const float* bk = (const float*)d_in[5];
  const float* wv = (const float*)d_in[6];
  const float* bv = (const float*)d_in[7];
  const float* wo = (const float*)d_in[8];
  const float* bo = (const float*)d_in[9];
  float* out = (float*)d_out;

  char* ws = (char*)d_ws;
  const size_t MB = 1024ull * 1024ull;
  _Float16* xh  = (_Float16*)(ws);            // 16MB, later reused as Vt
  _Float16* wqT = (_Float16*)(ws + 16 * MB);  //  8MB each
  _Float16* wkT = (_Float16*)(ws + 24 * MB);
  _Float16* wvT = (_Float16*)(ws + 32 * MB);
  _Float16* woT = (_Float16*)(ws + 40 * MB);
  _Float16* Qh  = (_Float16*)(ws + 48 * MB);  // 16MB each
  _Float16* Kh  = (_Float16*)(ws + 64 * MB);
  _Float16* Vh  = (_Float16*)(ws + 80 * MB);
  _Float16* Oh  = (_Float16*)(ws + 96 * MB);
  _Float16* Vt  = xh; // alias: xh's last use is the V projection GEMM

  cvt_f32_f16<<<(int)((NELEM + 255) / 256), 256, 0, stream>>>(x, xh, NELEM);
  long wN = (long)Dc * Dc;
  transpose_to_f16<<<(int)((wN + 255) / 256), 256, 0, stream>>>(wq, wqT, Dc);
  transpose_to_f16<<<(int)((wN + 255) / 256), 256, 0, stream>>>(wk, wkT, Dc);
  transpose_to_f16<<<(int)((wN + 255) / 256), 256, 0, stream>>>(wv, wvT, Dc);
  transpose_to_f16<<<(int)((wN + 255) / 256), 256, 0, stream>>>(wo, woT, Dc);

  dim3 gg(Dc / 256, (int)(NTOK / 64));   // (8, 64)
  gemm_f16_wmma<_Float16><<<gg, 256, 0, stream>>>(xh, wqT, bq, Qh, (int)NTOK, Dc, Dc);
  gemm_f16_wmma<_Float16><<<gg, 256, 0, stream>>>(xh, wkT, bk, Kh, (int)NTOK, Dc, Dc);
  gemm_f16_wmma<_Float16><<<gg, 256, 0, stream>>>(xh, wvT, bv, Vh, (int)NTOK, Dc, Dc);

  rope_inplace<<<(Bc * Sc * Hc) / 256, 256, 0, stream>>>(Qh, Kh);
  transpose_v_kernel<<<(int)((NELEM + 255) / 256), 256, 0, stream>>>(Vh, Vt);

  attn_wmma<<<(Bc * Hc * (Sc / 16)) / 4, 128, 0, stream>>>(Qh, Kh, Vt, am, Oh);

  gemm_f16_wmma<float><<<gg, 256, 0, stream>>>(Oh, woT, bo, out, (int)NTOK, Dc, Dc);
}
